// BottleneckBlock_37443524887216
// MI455X (gfx1250) — matrix operand order repaired and checked by reading, not compiled, not run on hardware
//
#include <hip/hip_runtime.h>

// ---------------------------------------------------------------------------
// VQ bottleneck for MI455X (gfx1250, wave32, WMMA).
//   x: (16, 64, 8192) f32,  k: (2048, 64) f32
//   dist[i,j] = ||x_i||^2 - 2 x_i.k_j + ||k_j||^2 ; argmin/min over j.
// Distance GEMM on v_wmma_f32_16x16x32_bf16 with A = bf16(-2*x), C = 0;
// ||k_j||^2 is folded into the argmin compare (post-WMMA v_add dual-issues
// with the cndmask stream and fills the WMMA->VALU hazard window).
// Each wave owns M=32 rows (two accumulator chains per B fragment pair);
// codebook chunks are double-buffered in LDS (prefetch in regs under compute).
// ---------------------------------------------------------------------------

#define T_DIM     8192
#define WIDTH     64
#define KBINS     2048
#define NROWS     131072            // 16 * 8192
#define NEL       8388608           // 131072 * 64
#define XL_OFF    0
#define XD_OFF    131072
#define SC_OFF    (131072 + 8388608)
#define CB_CHUNK  64                // codes per LDS chunk (double buffered)
#define NCHUNK    (KBINS / CB_CHUNK)
#define XT_STRIDE 72                // halves per xf-tile row (64 + pad)
#define CB_STRIDE 72                // halves per codebook row (64 + pad)

typedef __attribute__((ext_vector_type(16))) __bf16 v16bf;
typedef __attribute__((ext_vector_type(8)))  float  v8f;

union Frag { int4 i2[2]; v16bf v; };

// pack two 16B-aligned groups of 8 halves into one WMMA fragment
__device__ __forceinline__ v16bf pack2(const unsigned short* p0,
                                       const unsigned short* p1) {
  Frag f;
  f.i2[0] = *(const int4*)p0;
  f.i2[1] = *(const int4*)p1;
  return f.v;
}

// float -> bf16 (round to nearest even)
__device__ __forceinline__ unsigned short f2bf(float f) {
  unsigned int u = __float_as_uint(f);
  u = (u + 0x7FFFu + ((u >> 16) & 1u)) >> 16;
  return (unsigned short)u;
}

// ---------------------------------------------------------------------------
// Prep: bf16 codebook + ||k_j||^2 into workspace (done once per launch).
// ---------------------------------------------------------------------------
__global__ void vq_prep(const float* __restrict__ k,
                        float* __restrict__ kkws,
                        unsigned short* __restrict__ cbws) {
  int j = blockIdx.x * blockDim.x + threadIdx.x;
  if (j >= KBINS) return;
  const float* kr = k + (size_t)j * WIDTH;
  float s = 0.f;
#pragma unroll
  for (int c = 0; c < WIDTH; ++c) {
    float v = kr[c];
    s += v * v;
    cbws[(size_t)j * WIDTH + c] = f2bf(v);
  }
  kkws[j] = s;
}

// ---------------------------------------------------------------------------
// Main: 512 blocks x 256 threads; each wave handles 32 xf rows.
// ---------------------------------------------------------------------------
__launch_bounds__(256)
__global__ void vq_main(const float* __restrict__ x,
                        const float* __restrict__ k,
                        const float* __restrict__ kkws,
                        const unsigned short* __restrict__ cbws,
                        float* __restrict__ out,
                        float* __restrict__ acc) {
  __shared__ unsigned short xt[8][32 * XT_STRIDE];       // bf16(-2x), per wave
  __shared__ unsigned short cb[2][CB_CHUNK * CB_STRIDE]; // double-buffered chunk
  __shared__ float          kkl[2][CB_CHUNK];            // ||k||^2 chunk
  __shared__ int            bidx[8][32];                 // winning code per row

  const int tid  = threadIdx.x;
  const int w    = tid >> 5;
  const int lane = tid & 31;
  const int m    = lane & 15;   // A row within 16 / B col within 16
  const int g    = lane >> 4;   // lane group

  const int tileRow = blockIdx.x * 256 + w * 32;  // first xf row of this wave
  const int n  = tileRow >> 13;                   // /8192
  const int t0 = tileRow & (T_DIM - 1);

  // ---- stage A tile (transpose x -> xf), accumulate stats -----------------
  float sx = 0.f, sx2 = 0.f;
  float xk0[8], xk1[8];                  // ||x_row||^2 for this lane's rows
  {
    const float* xn = x + (size_t)n * WIDTH * T_DIM + t0;
#pragma unroll
    for (int h = 0; h < 2; ++h) {        // row halves 0..15, 16..31
      float s[16];
#pragma unroll
      for (int t = 0; t < 16; ++t) s[t] = 0.f;
#pragma unroll
      for (int rep = 0; rep < 2; ++rep) {
        const int c = lane + 32 * rep;
        const float* p = xn + (size_t)c * T_DIM + h * 16;
#pragma unroll
        for (int q = 0; q < 4; ++q) {
          float4 v = *(const float4*)(p + q * 4);
          float vv[4] = { v.x, v.y, v.z, v.w };
#pragma unroll
          for (int e = 0; e < 4; ++e) {
            const int t = q * 4 + e;
            const float f = vv[e];
            sx  += f;
            sx2 += f * f;
            s[t] += f * f;
            xt[w][(h * 16 + t) * XT_STRIDE + c] = f2bf(-2.0f * f);
          }
        }
      }
#pragma unroll
      for (int off = 16; off >= 1; off >>= 1) {
#pragma unroll
        for (int t = 0; t < 16; ++t) s[t] += __shfl_xor(s[t], off, 32);
      }
#pragma unroll
      for (int r = 0; r < 8; ++r) {
        const float v = (g == 0) ? s[r] : s[r + 8];  // rows g*8+r of this half
        if (h == 0) xk0[r] = v; else xk1[r] = v;
      }
    }
  }
#pragma unroll
  for (int off = 16; off >= 1; off >>= 1) {
    sx  += __shfl_xor(sx,  off, 32);
    sx2 += __shfl_xor(sx2, off, 32);
  }
  if (lane == 0) {
    atomicAdd(acc + 0, sx);
    atomicAdd(acc + 1, sx2);
  }

  // ---- prologue: chunk 0 into buffer 0 ------------------------------------
  {
    const int4* src = (const int4*)cbws;      // 64 rows * 8 int4 = 512 int4
    const int i0 = 2 * tid, i1 = i0 + 1;
    int4 p0 = src[i0], p1 = src[i1];
    *(int4*)&cb[0][(i0 >> 3) * CB_STRIDE + (i0 & 7) * 8] = p0;
    *(int4*)&cb[0][(i1 >> 3) * CB_STRIDE + (i1 & 7) * 8] = p1;
    if (tid < CB_CHUNK) kkl[0][tid] = kkws[tid];
  }
  __syncthreads();   // also covers xt staging -> fragment reads

  // ---- A fragments: rows m (tile0) and 16+m (tile1), K halves 0/1 ---------
  const unsigned short* arow0 = &xt[w][m * XT_STRIDE];
  const unsigned short* arow1 = &xt[w][(16 + m) * XT_STRIDE];
  const v16bf a0 = pack2(arow0 + g * 8,      arow0 + 16 + g * 8);
  const v16bf a1 = pack2(arow0 + 32 + g * 8, arow0 + 48 + g * 8);
  const v16bf a2 = pack2(arow1 + g * 8,      arow1 + 16 + g * 8);
  const v16bf a3 = pack2(arow1 + 32 + g * 8, arow1 + 48 + g * 8);

  const v8f vzero = { 0.f, 0.f, 0.f, 0.f, 0.f, 0.f, 0.f, 0.f };

  float best0[8], best1[8];
  int   bj0[8],   bj1[8];
#pragma unroll
  for (int r = 0; r < 8; ++r) {
    best0[r] = 3.0e38f; best1[r] = 3.0e38f; bj0[r] = 0; bj1[r] = 0;
  }

  // ---- codebook chunk loop (double buffered) ------------------------------
  for (int ch = 0; ch < NCHUNK; ++ch) {
    const int cur = ch & 1;

    // prefetch next chunk into registers (latency hidden by WMMA below)
    int4 p0, p1; float pkk = 0.f;
    if (ch < NCHUNK - 1) {
      const int4* src = (const int4*)(cbws + (size_t)(ch + 1) * CB_CHUNK * WIDTH);
      p0 = src[2 * tid];
      p1 = src[2 * tid + 1];
      if (tid < CB_CHUNK) pkk = kkws[(ch + 1) * CB_CHUNK + tid];
    }

#pragma unroll
    for (int jb = 0; jb < CB_CHUNK / 16; ++jb) {
      const unsigned short* bp = &cb[cur][(jb * 16 + m) * CB_STRIDE + g * 8];
      const v16bf b0 = pack2(bp,      bp + 16);  // K 0..31
      const v16bf b1 = pack2(bp + 32, bp + 48);  // K 32..63
      const float kkv = kkl[cur][jb * 16 + m];

      v8f c0 = __builtin_amdgcn_wmma_f32_16x16x32_bf16(
                   false, a0, false, b0, (short)0, vzero, false, false);
      v8f c1 = __builtin_amdgcn_wmma_f32_16x16x32_bf16(
                   false, a2, false, b0, (short)0, vzero, false, false);
      c0 = __builtin_amdgcn_wmma_f32_16x16x32_bf16(
                   false, a1, false, b1, (short)0, c0, false, false);
      c1 = __builtin_amdgcn_wmma_f32_16x16x32_bf16(
                   false, a3, false, b1, (short)0, c1, false, false);

      const int jg = ch * CB_CHUNK + jb * 16 + m;
#pragma unroll
      for (int r = 0; r < 8; ++r) {
        const float d0 = c0[r] + kkv;           // + ||k_j||^2
        if (d0 < best0[r]) { best0[r] = d0; bj0[r] = jg; }
        const float d1 = c1[r] + kkv;
        if (d1 < best1[r]) { best1[r] = d1; bj1[r] = jg; }
      }
    }

    // commit prefetched chunk into the spare buffer
    if (ch < NCHUNK - 1) {
      const int nxt = cur ^ 1;
      const int i0 = 2 * tid, i1 = i0 + 1;
      *(int4*)&cb[nxt][(i0 >> 3) * CB_STRIDE + (i0 & 7) * 8] = p0;
      *(int4*)&cb[nxt][(i1 >> 3) * CB_STRIDE + (i1 & 7) * 8] = p1;
      if (tid < CB_CHUNK) kkl[nxt][tid] = pkk;
    }
    __syncthreads();
  }

  // ---- argmin across the 16-lane group (each lane covers j % 16 == m) -----
#pragma unroll
  for (int off = 8; off >= 1; off >>= 1) {
#pragma unroll
    for (int r = 0; r < 8; ++r) {
      float ob; int oj;
      ob = __shfl_xor(best0[r], off, 16);
      oj = __shfl_xor(bj0[r],   off, 16);
      if (ob < best0[r] || (ob == best0[r] && oj < bj0[r])) {
        best0[r] = ob; bj0[r] = oj;
      }
      ob = __shfl_xor(best1[r], off, 16);
      oj = __shfl_xor(bj1[r],   off, 16);
      if (ob < best1[r] || (ob == best1[r] && oj < bj1[r])) {
        best1[r] = ob; bj1[r] = oj;
      }
    }
  }

  if (m == 0) {                  // lanes 0 and 16
    float localS = 0.f;
#pragma unroll
    for (int r = 0; r < 8; ++r) {
      const int row0 = g * 8 + r;         // rows 0..15
      const int row1 = 16 + g * 8 + r;    // rows 16..31
      localS += best0[r] + xk0[r];
      localS += best1[r] + xk1[r];
      out[XL_OFF + tileRow + row0] = (float)bj0[r];
      out[XL_OFF + tileRow + row1] = (float)bj1[r];
      bidx[w][row0] = bj0[r];
      bidx[w][row1] = bj1[r];
    }
    atomicAdd(acc + 2, localS);
  }
  __syncthreads();

  // ---- x_d = k[x_l] gather, written back in (N, width, T) layout ----------
  {
    float* od = out + XD_OFF + (size_t)n * WIDTH * T_DIM + t0;
#pragma unroll 4
    for (int it = 0; it < 64; ++it) {
      const int idx = it * 32 + lane;
      const int c = idx >> 5, t = idx & 31;   // 32 consecutive t per wave-iter
      const int j = bidx[w][t];
      od[(size_t)c * T_DIM + t] = k[(size_t)j * WIDTH + c];
    }
  }
}

// ---------------------------------------------------------------------------
// Finalize scalars.
// ---------------------------------------------------------------------------
__global__ void vq_finalize(const float* __restrict__ acc,
                            float* __restrict__ out) {
  if (threadIdx.x == 0 && blockIdx.x == 0) {
    const float sx  = acc[0];
    const float sx2 = acc[1];
    const float sm  = acc[2];
    const float nel = (float)NEL;
    const float var = (sx2 - sx * sx / nel) / nel;
    out[SC_OFF + 0] = sm / nel;                // commit_loss
    out[SC_OFF + 1] = sm / (float)NROWS;       // fit
    out[SC_OFF + 2] = sqrtf(fmaxf(var, 0.f));  // prenorm
  }
}

// ---------------------------------------------------------------------------
extern "C" void kernel_launch(void* const* d_in, const int* in_sizes, int n_in,
                              void* d_out, int out_size, void* d_ws, size_t ws_size,
                              hipStream_t stream) {
  (void)in_sizes; (void)n_in; (void)out_size; (void)ws_size;
  const float* x = (const float*)d_in[0];
  const float* k = (const float*)d_in[1];
  float* out = (float*)d_out;
  float* ws  = (float*)d_ws;

  // workspace layout (floats): [0..15] accumulators, [16..2063] ||k||^2,
  // then bf16 codebook (2048*64 halves, 16B aligned at float index 2064).
  float* accp = ws;
  float* kkws = ws + 16;
  unsigned short* cbws = (unsigned short*)(ws + 16 + KBINS);

  hipMemsetAsync(accp, 0, 16 * sizeof(float), stream);
  vq_prep<<<KBINS / 256, 256, 0, stream>>>(k, kkws, cbws);
  vq_main<<<NROWS / 256, 256, 0, stream>>>(x, k, kkws, cbws, out, accp);
  vq_finalize<<<1, 32, 0, stream>>>(accp, out);
}
